// AnchorTransformer_22265110462889
// MI455X (gfx1250) — compile-verified, matrix operand order checked by reading, not run on hardware
//
#include <hip/hip_runtime.h>
#include <hip/hip_bf16.h>

typedef __attribute__((ext_vector_type(16))) _Float16 v16h;
typedef __attribute__((ext_vector_type(8)))  float    v8f;

#define C_DIM 256

union FragH {
    v16h v;
    _Float16 h[16];
    unsigned int u[8];
};

// K-index inside a 16x32 16-bit WMMA fragment (ISA 7.12.2):
// lanes 0-15 hold K = {0..7, 16..23}, lanes 16-31 hold K = {8..15, 24..31};
// VGPR j holds the pair (k, k+1).
__device__ __forceinline__ int frag_k(int j, int hi8) {
    return ((j >> 2) << 4) + hi8 + ((j & 3) << 1);
}

// Load a 16x32 fragment from an f32 row-major row, converting to f16.
__device__ __forceinline__ v16h load_frag_f32(const float* __restrict__ row, int k0, int hi8) {
    FragH f;
#pragma unroll
    for (int j = 0; j < 8; ++j) {
        int kk = k0 + frag_k(j, hi8);
        float2 p = *(const float2*)(row + kk);
        f.h[2 * j]     = (_Float16)p.x;
        f.h[2 * j + 1] = (_Float16)p.y;
    }
    return f.v;
}

// Load a 16x32 fragment from an f16 row-major row (dword per pair).
__device__ __forceinline__ v16h load_frag_f16(const _Float16* __restrict__ row, int k0, int hi8) {
    FragH f;
#pragma unroll
    for (int j = 0; j < 8; ++j) {
        int kk = k0 + frag_k(j, hi8);
        f.u[j] = *(const unsigned int*)(row + kk);
    }
    return f.v;
}

// D = A(f32, MxC) * W(f32 rows = output dims, offset woff)^T + bias
// Each wave: 16(M) x 64(N) strip. 8 waves/block -> 128 x 64 block tile.
// grid.x = M/128, grid.y = 256/64 = 4.
template <bool OUT_HALF>
__global__ __launch_bounds__(256) void gemm256_kernel(
    const float* __restrict__ A,
    const float* __restrict__ W, int woff,
    const float* __restrict__ bias, int boff,
    void* __restrict__ out)
{
    const int lane  = threadIdx.x & 31;
    const int wave  = threadIdx.x >> 5;
    const int row16 = lane & 15;
    const int hi8   = ((lane >> 4) & 1) << 3;   // 0 or 8
    const int m0    = blockIdx.x * 128 + wave * 16;
    const int n0    = blockIdx.y * 64;

    const float* arow = A + (size_t)(m0 + row16) * C_DIM;

    v8f acc[4] = {};   // f32 accumulators, 4 n-subtiles of 16

    for (int k0 = 0; k0 < C_DIM; k0 += 32) {
        v16h a = load_frag_f32(arow, k0, hi8);
#pragma unroll
        for (int t = 0; t < 4; ++t) {
            const float* wrow = W + (size_t)(woff + n0 + t * 16 + row16) * C_DIM;
            v16h b = load_frag_f32(wrow, k0, hi8);
            acc[t] = __builtin_amdgcn_wmma_f32_16x16x32_f16(
                false, a, false, b, (short)0, acc[t], false, false);
        }
    }

    // C/D layout: lanes 0-15 -> M = r, N = lane; lanes 16-31 -> M = 8 + r, N = lane - 16
#pragma unroll
    for (int t = 0; t < 4; ++t) {
        const int n = n0 + t * 16 + row16;
        const float bv = bias[boff + n];
#pragma unroll
        for (int r = 0; r < 8; ++r) {
            const int m = m0 + hi8 + r;
            const float val = acc[t][r] + bv;
            if (OUT_HALF)
                ((_Float16*)out)[(size_t)m * C_DIM + n] = (_Float16)val;
            else
                ((float*)out)[(size_t)m * C_DIM + n] = val;
        }
    }
}

// Final GEMM: o2 = o(f16) @ out_w^T + out_b, fused epilogue:
// out = feat + (lab > 0 ? o2 : 0)
__global__ __launch_bounds__(256) void gemm_out_kernel(
    const _Float16* __restrict__ A,
    const float* __restrict__ W,
    const float* __restrict__ bias,
    const float* __restrict__ feat,
    const int* __restrict__ lab,
    float* __restrict__ out)
{
    const int lane  = threadIdx.x & 31;
    const int wave  = threadIdx.x >> 5;
    const int row16 = lane & 15;
    const int hi8   = ((lane >> 4) & 1) << 3;
    const int m0    = blockIdx.x * 128 + wave * 16;
    const int n0    = blockIdx.y * 64;

    const _Float16* arow = A + (size_t)(m0 + row16) * C_DIM;

    v8f acc[4] = {};

    for (int k0 = 0; k0 < C_DIM; k0 += 32) {
        v16h a = load_frag_f16(arow, k0, hi8);
#pragma unroll
        for (int t = 0; t < 4; ++t) {
            const float* wrow = W + (size_t)(n0 + t * 16 + row16) * C_DIM;
            v16h b = load_frag_f32(wrow, k0, hi8);
            acc[t] = __builtin_amdgcn_wmma_f32_16x16x32_f16(
                false, a, false, b, (short)0, acc[t], false, false);
        }
    }

#pragma unroll
    for (int r = 0; r < 8; ++r) {
        const int m = m0 + hi8 + r;
        const bool keep = lab[m] > 0;
#pragma unroll
        for (int t = 0; t < 4; ++t) {
            const int n = n0 + t * 16 + row16;
            const float o2 = acc[t][r] + bias[n];
            const float f  = feat[(size_t)m * C_DIM + n];
            out[(size_t)m * C_DIM + n] = f + (keep ? o2 : 0.0f);
        }
    }
}

// One wave per pixel: gather K[idx], V[idx] (L2-resident, 1 MB total),
// scores via xor-butterfly wave reductions, f32 softmax, weighted V sum.
__global__ __launch_bounds__(256) void attn_kernel(
    const _Float16* __restrict__ qbuf,
    const float* __restrict__ Kbuf,
    const float* __restrict__ Vbuf,
    const int* __restrict__ lab,
    _Float16* __restrict__ obuf)
{
    const int L = 8;
    const int lane = threadIdx.x & 31;
    const int pix  = blockIdx.x * 8 + (threadIdx.x >> 5);

    const int l   = lab[pix];
    const int idx = (l > 0) ? (l - 1) : 0;

    const _Float16* qp = qbuf + (size_t)pix * C_DIM;
    float qr[8];
#pragma unroll
    for (int i = 0; i < 8; ++i) qr[i] = (float)qp[lane + 32 * i];

    const float* Kp = Kbuf + (size_t)idx * (L * C_DIM);
    const float* Vp = Vbuf + (size_t)idx * (L * C_DIM);

    const float scale = 0.0625f;   // 1/sqrt(256)
    float sc[L];
#pragma unroll
    for (int a = 0; a < L; ++a) {
        float s = 0.0f;
#pragma unroll
        for (int i = 0; i < 8; ++i)
            s += qr[i] * Kp[a * C_DIM + lane + 32 * i];
#pragma unroll
        for (int off = 16; off > 0; off >>= 1)
            s += __shfl_xor(s, off, 32);
        sc[a] = s * scale;
    }

    float mx = sc[0];
#pragma unroll
    for (int a = 1; a < L; ++a) mx = fmaxf(mx, sc[a]);
    float sum = 0.0f;
#pragma unroll
    for (int a = 0; a < L; ++a) { sc[a] = __expf(sc[a] - mx); sum += sc[a]; }
    const float inv = 1.0f / sum;
#pragma unroll
    for (int a = 0; a < L; ++a) sc[a] *= inv;

    _Float16* op = obuf + (size_t)pix * C_DIM;
#pragma unroll
    for (int i = 0; i < 8; ++i) {
        float acc = 0.0f;
#pragma unroll
        for (int a = 0; a < L; ++a)
            acc += sc[a] * Vp[a * C_DIM + lane + 32 * i];
        op[lane + 32 * i] = (_Float16)acc;
    }
}

extern "C" void kernel_launch(void* const* d_in, const int* in_sizes, int n_in,
                              void* d_out, int out_size, void* d_ws, size_t ws_size,
                              hipStream_t stream)
{
    const float* anchors   = (const float*)d_in[0];   // 64*8*256
    const float* features  = (const float*)d_in[1];   // 8*64*64*256
    const int*   inst      = (const int*)d_in[2];     // 8*64*64
    const float* in_proj_w = (const float*)d_in[3];   // 768*256
    const float* in_proj_b = (const float*)d_in[4];   // 768
    const float* out_w     = (const float*)d_in[5];   // 256*256
    const float* out_b     = (const float*)d_in[6];   // 256
    float* out = (float*)d_out;

    const int NPIX = 8 * 64 * 64;   // 32768
    const int MKV  = 64 * 8;        // 512

    char* ws = (char*)d_ws;
    float* Kbuf = (float*)ws;            ws += (size_t)MKV * C_DIM * sizeof(float);
    float* Vbuf = (float*)ws;            ws += (size_t)MKV * C_DIM * sizeof(float);
    _Float16* qbuf = (_Float16*)ws;      ws += (size_t)NPIX * C_DIM * sizeof(_Float16);
    _Float16* obuf = (_Float16*)ws;

    dim3 blk(256);

    // K = anchors @ Wk^T + bk ; V = anchors @ Wv^T + bv   (M = 512 -> grid 4x4)
    gemm256_kernel<false><<<dim3(4, 4), blk, 0, stream>>>(
        anchors, in_proj_w, C_DIM, in_proj_b, C_DIM, (void*)Kbuf);
    gemm256_kernel<false><<<dim3(4, 4), blk, 0, stream>>>(
        anchors, in_proj_w, 2 * C_DIM, in_proj_b, 2 * C_DIM, (void*)Vbuf);

    // q = features @ Wq^T + bq   (M = 32768 -> grid 256x4), f16 output
    gemm256_kernel<true><<<dim3(NPIX / 128, 4), blk, 0, stream>>>(
        features, in_proj_w, 0, in_proj_b, 0, (void*)qbuf);

    // gathered attention, one wave per pixel
    attn_kernel<<<dim3(NPIX / 8), blk, 0, stream>>>(qbuf, Kbuf, Vbuf, inst, obuf);

    // out = feat + mask * (o @ out_w^T + out_b)
    gemm_out_kernel<<<dim3(NPIX / 128, 4), blk, 0, stream>>>(
        obuf, out_w, out_b, features, inst, out);
}
